// NonLocalBlock1D_position_5257039970793
// MI455X (gfx1250) — matrix operand order repaired and checked by reading, not compiled, not run on hardware
//
#include <hip/hip_runtime.h>
#include <hip/hip_bf16.h>

// ---------------- problem constants ----------------
#define Bn 8
#define Cn 512
#define Tn 2048
#define Dn 256
#define EPSn 1e-5f

typedef __bf16 bf16;
typedef bf16  v16bf __attribute__((ext_vector_type(16)));
typedef bf16  v8bf  __attribute__((ext_vector_type(8)));
typedef float v8f   __attribute__((ext_vector_type(8)));

// ---------------- WMMA fragment helpers (wave32, 16x16x32 bf16) ----------------
// A fragment: 16x32 (MxK) bf16, row-major src. Lane half h holds K = 16h..16h+15
// (one contiguous 16-elem K run).
__device__ __forceinline__ v16bf load_a_frag(const bf16* src, int lda) {
    int lane = threadIdx.x & 31;
    int m = lane & 15, h = lane >> 4;
    v16bf a;
#pragma unroll
    for (int e = 0; e < 16; ++e) {
        int k = 16 * h + e;
        a[e] = src[m * lda + k];
    }
    return a;
}

// Swizzled fragment: one 32x16 B-tile stored as 32 lanes x 16 contiguous bf16.
__device__ __forceinline__ v16bf load_frag_sw(const bf16* base) {
    int lane = threadIdx.x & 31;
    return *(const v16bf*)(base + lane * 16);
}

// Swizzle index for storing B(k, n), k in [0,32): lane=(k>>4)*16+(n&15), elem=k&15.
__device__ __forceinline__ int b_sw_idx(int k, int n) {
    return ((n >> 4) << 9) + (((((k >> 4) << 4) + (n & 15))) << 4) + (k & 15);
}

__device__ __forceinline__ v8f wmma_bf16(v16bf a, v16bf b, v8f c) {
    return __builtin_amdgcn_wmma_f32_16x16x32_bf16(false, a, false, b, (short)0, c,
                                                   false, false);
}

// ---------------- small prep kernels ----------------
__global__ void cvt_f32_bf16_kernel(const float* __restrict__ src,
                                    bf16* __restrict__ dst, int n) {
    int i = blockIdx.x * blockDim.x + threadIdx.x;
    if (i < n) dst[i] = (bf16)src[i];
}

// Collapse {identity, conv d=1, conv d=2} + z-conv into 5 shift matrices M_s.
__global__ void combine_w_kernel(const float* __restrict__ zw,   // (C, 3C)
                                 const float* __restrict__ w1,   // (C,1,3,C)
                                 const float* __restrict__ w2,   // (C,1,3,C)
                                 bf16* __restrict__ Mall) {
    int i = blockIdx.x * blockDim.x + threadIdx.x;
    if (i >= 5 * Cn * Cn) return;
    int c = i & (Cn - 1);
    int o = (i >> 9) & (Cn - 1);
    int s = i >> 18;
    const float* zr = zw + (size_t)o * (3 * Cn);
    float acc = 0.f;
    if (s == 2) {
        acc = zr[c];
        for (int op = 0; op < Cn; ++op) {
            acc += zr[Cn + op]     * w1[(size_t)op * (3 * Cn) + Cn + c];
            acc += zr[2 * Cn + op] * w2[(size_t)op * (3 * Cn) + Cn + c];
        }
    } else {
        const float* w = (s == 0 || s == 4) ? w2 : w1;
        const float* zpart = zr + ((s == 0 || s == 4) ? 2 * Cn : Cn);
        int k = (s < 2) ? 0 : 2;
        for (int op = 0; op < Cn; ++op)
            acc += zpart[op] * w[(size_t)op * (3 * Cn) + k * Cn + c];
    }
    Mall[i] = (bf16)acc;
}

// x_pe[b,c,t] = bf16( x[b,c,t] + PE[t,c] )
__global__ void pe_add_kernel(const float* __restrict__ x, bf16* __restrict__ xpe) {
    size_t i = (size_t)blockIdx.x * blockDim.x + threadIdx.x;
    if (i >= (size_t)Bn * Cn * Tn) return;
    int t = (int)(i % Tn);
    int c = (int)((i / Tn) % Cn);
    float div = __expf(-9.210340371976184f * (float)(c & ~1) / (float)Cn);
    float arg = (float)t * div;
    float pe = (c & 1) ? __cosf(arg) : __sinf(arg);
    xpe[i] = (bf16)(x[i] + pe);
}

// ---------------- staging helpers ----------------
__device__ __forceinline__ void stage_convz(bf16* At, bf16* Bt,
                                            const bf16* __restrict__ Mall,
                                            const bf16* __restrict__ xb,
                                            int kk, int row0, int col0, int tid) {
    const int s = kk >> 9, c0 = kk & (Cn - 1), shift = s - 2;
    const bf16* Ag = Mall + (size_t)s * Cn * Cn + (size_t)row0 * Cn + c0;
    for (int i = tid; i < 512; i += 256) {
        int r = i >> 2, c8 = (i & 3) * 8;
        *(v8bf*)(At + r * 32 + c8) = *(const v8bf*)(Ag + (size_t)r * Cn + c8);
    }
    for (int i = tid; i < 32 * 128; i += 256) {
        int r = i >> 7, c = i & 127;
        int t = col0 + c + shift;
        bf16 v = (t >= 0 && t < Tn) ? xb[(size_t)(c0 + r) * Tn + t] : (bf16)0.f;
        Bt[b_sw_idx(r, c)] = v;
    }
}

__device__ __forceinline__ void stage_rowmajorB(bf16* At, bf16* Bt,
                                                const bf16* __restrict__ Ag,
                                                const bf16* __restrict__ Bg,
                                                int ldb, int tid) {
    // A: 128x32 row-major vector copy; B is not staged (Bg, ldb unused)
    for (int i = tid; i < 512; i += 256) {
        int r = i >> 2, c8 = (i & 3) * 8;
        *(v8bf*)(At + r * 32 + c8) = *(const v8bf*)(Ag + (size_t)r * 32 /*lda? no*/ + c8);
    }
    (void)Bg; (void)ldb;
}

// ---------------- GEMM-1: x2 = sum_s M_s @ shift(x_pe,s) + z_b ----------------
// grid: (T/128, C/128, B), block 256 (8 waves). Double-buffered LDS.
__global__ void conv_z_gemm_kernel(const bf16* __restrict__ Mall,
                                   const bf16* __restrict__ xpe,
                                   const float* __restrict__ z_b,
                                   float* __restrict__ x2f,
                                   bf16* __restrict__ x2h) {
    __shared__ __align__(32) bf16 At[2][128 * 32];
    __shared__ __align__(32) bf16 Bt[2][32 * 128];
    const int tn = blockIdx.x, tm = blockIdx.y, b = blockIdx.z;
    const int tid = threadIdx.x, wave = tid >> 5, lane = tid & 31;
    const int col0 = tn * 128, row0 = tm * 128;
    const bf16* xb = xpe + (size_t)b * Cn * Tn;

    v8f acc[8] = {};
    int p = 0;
    stage_convz(At[0], Bt[0], Mall, xb, 0, row0, col0, tid);
    for (int kk = 0; kk < 5 * Cn; kk += 32) {
        __syncthreads();
        if (kk + 32 < 5 * Cn)
            stage_convz(At[p ^ 1], Bt[p ^ 1], Mall, xb, kk + 32, row0, col0, tid);
        v16bf a = load_a_frag(At[p] + wave * 16 * 32, 32);
        v16bf bfr[8];
#pragma unroll
        for (int j = 0; j < 8; ++j) bfr[j] = load_frag_sw(Bt[p] + j * 512);
#pragma unroll
        for (int j = 0; j < 8; ++j) acc[j] = wmma_bf16(a, bfr[j], acc[j]);
        p ^= 1;
    }
    const int h = lane >> 4, n = lane & 15;
#pragma unroll
    for (int j = 0; j < 8; ++j)
#pragma unroll
        for (int r = 0; r < 8; ++r) {
            int row = row0 + wave * 16 + r + 8 * h;
            int col = col0 + j * 16 + n;
            float v = acc[j][r] + z_b[row];
            size_t idx = ((size_t)b * Cn + row) * Tn + col;
            x2f[idx] = v;
            x2h[idx] = (bf16)v;
        }
}

// ---------------- GEMM-2: theta/phi/g projections ----------------
// grid: (T/128, D/128, 3*B). which: 0=theta ((T,D)), 1=phi ((D,T)), 2=g ((T,D)).
__global__ void proj_gemm_kernel(const bf16* __restrict__ wproj,  // [3][D][C]
                                 const float* __restrict__ th_b,
                                 const float* __restrict__ ph_b,
                                 const float* __restrict__ g_b,
                                 const bf16* __restrict__ x2h,
                                 bf16* __restrict__ theta,  // (B,T,D)
                                 bf16* __restrict__ phi,    // (B,D,T)
                                 bf16* __restrict__ gx) {   // (B,T,D)
    __shared__ __align__(32) bf16 At[2][128 * 32];
    __shared__ __align__(32) bf16 Bt[2][32 * 128];
    const int tn = blockIdx.x, tm = blockIdx.y;
    const int which = blockIdx.z >> 3, b = blockIdx.z & 7;
    const int tid = threadIdx.x, wave = tid >> 5, lane = tid & 31;
    const int col0 = tn * 128, row0 = tm * 128;
    const bf16* W = wproj + (size_t)which * Dn * Cn;
    const float* bias = (which == 0) ? th_b : (which == 1) ? ph_b : g_b;

    v8f acc[8] = {};
    int p = 0;
    auto stage = [&](int buf, int k0) {
        const bf16* Ag = W + (size_t)row0 * Cn + k0;
        for (int i = tid; i < 512; i += 256) {
            int r = i >> 2, c8 = (i & 3) * 8;
            *(v8bf*)(At[buf] + r * 32 + c8) = *(const v8bf*)(Ag + (size_t)r * Cn + c8);
        }
        const bf16* Bg = x2h + ((size_t)b * Cn + k0) * Tn + col0;
        for (int i = tid; i < 512; i += 256) {
            int r = i >> 4, c8 = (i & 15) * 8;
            v8bf v = *(const v8bf*)(Bg + (size_t)r * Tn + c8);
#pragma unroll
            for (int e = 0; e < 8; ++e) Bt[buf][b_sw_idx(r, c8 + e)] = v[e];
        }
    };
    stage(0, 0);
    for (int k0 = 0; k0 < Cn; k0 += 32) {
        __syncthreads();
        if (k0 + 32 < Cn) stage(p ^ 1, k0 + 32);
        v16bf a = load_a_frag(At[p] + wave * 16 * 32, 32);
        v16bf bfr[8];
#pragma unroll
        for (int j = 0; j < 8; ++j) bfr[j] = load_frag_sw(Bt[p] + j * 512);
#pragma unroll
        for (int j = 0; j < 8; ++j) acc[j] = wmma_bf16(a, bfr[j], acc[j]);
        p ^= 1;
    }
    const int h = lane >> 4, n = lane & 15;
#pragma unroll
    for (int j = 0; j < 8; ++j)
#pragma unroll
        for (int r = 0; r < 8; ++r) {
            int row = row0 + wave * 16 + r + 8 * h;   // d
            int col = col0 + j * 16 + n;              // t
            float v = acc[j][r] + bias[row];
            if (which == 1) {
                phi[((size_t)b * Dn + row) * Tn + col] = (bf16)v;
            } else {
                bf16* out = (which == 0) ? theta : gx;
                out[((size_t)b * Tn + col) * Dn + row] = (bf16)v;
            }
        }
}

// ---------------- flash attention: y = softmax(theta @ phi) @ g ----------------
// grid: (T/128, B), block 256 (8 waves). Double-buffered phi/g panels; Q in regs;
// online softmax via half-wave shfl; P routed through wave-private LDS.
__global__ void flash_attn_kernel(const bf16* __restrict__ theta,  // (B,T,D)
                                  const bf16* __restrict__ phi,    // (B,D,T)
                                  const bf16* __restrict__ gx,     // (B,T,D)
                                  bf16* __restrict__ yout) {       // (B,T,D)
    __shared__ __align__(32) bf16 phiT[2][Dn * 32];
    __shared__ __align__(32) bf16 gT[2][32 * Dn];
    __shared__ __align__(32) bf16 pStage[8 * 16 * 32];
    const int b = blockIdx.y;
    const int tid = threadIdx.x, wave = tid >> 5, lane = tid & 31;
    const int qrow0 = blockIdx.x * 128 + wave * 16;
    const int h = lane >> 4, n = lane & 15;

    v16bf qf[8];
    const bf16* thb = theta + ((size_t)b * Tn + qrow0) * Dn;
#pragma unroll
    for (int kk = 0; kk < 8; ++kk) qf[kk] = load_a_frag(thb + kk * 32, Dn);

    v8f yacc[16] = {};
    float mstate[8], lstate[8];
#pragma unroll
    for (int r = 0; r < 8; ++r) { mstate[r] = -1e30f; lstate[r] = 0.f; }

    auto stage = [&](int buf, int cb) {
        for (int i = tid; i < 1024; i += 256) {
            int d = i >> 2, c8 = (i & 3) * 8;
            v8bf v = *(const v8bf*)(phi + ((size_t)b * Dn + d) * Tn + cb + c8);
#pragma unroll
            for (int e = 0; e < 8; ++e) {
                int c = c8 + e;
                phiT[buf][(((d >> 5) * 2 + (c >> 4)) << 9) +
                          (((((d >> 4) & 1) << 4) + (c & 15)) << 4) + (d & 15)] = v[e];
            }
        }
        for (int i = tid; i < 1024; i += 256) {
            int r = i >> 5, c8 = (i & 31) * 8;
            v8bf v = *(const v8bf*)(gx + ((size_t)b * Tn + cb + r) * Dn + c8);
#pragma unroll
            for (int e = 0; e < 8; ++e) gT[buf][b_sw_idx(r, c8 + e)] = v[e];
        }
    };

    int p = 0;
    stage(0, 0);
    for (int cb = 0; cb < Tn; cb += 32) {
        __syncthreads();
        if (cb + 32 < Tn) stage(p ^ 1, cb + 32);
        const bf16* ph = phiT[p];
        const bf16* gp = gT[p];

        // S (16 x 32) = Q @ phi_panel, phi fragments pipelined one step ahead
        v8f s0 = {}, s1 = {};
        v16bf cb0 = load_frag_sw(ph + 0);
        v16bf cb1 = load_frag_sw(ph + 512);
#pragma unroll
        for (int kk = 0; kk < 8; ++kk) {
            v16bf nb0 = cb0, nb1 = cb1;
            if (kk < 7) {
                nb0 = load_frag_sw(ph + (kk * 2 + 2) * 512);
                nb1 = load_frag_sw(ph + (kk * 2 + 3) * 512);
            }
            s0 = wmma_bf16(qf[kk], cb0, s0);
            s1 = wmma_bf16(qf[kk], cb1, s1);
            cb0 = nb0; cb1 = nb1;
        }

        // online softmax (row r of this lane's half covers query row r+8h)
        float cf[8];
#pragma unroll
        for (int r = 0; r < 8; ++r) {
            float mx = fmaxf(s0[r], s1[r]);
#pragma unroll
            for (int m = 1; m < 16; m <<= 1) mx = fmaxf(mx, __shfl_xor(mx, m, 32));
            float mnew = fmaxf(mstate[r], mx);
            float p0 = __expf(s0[r] - mnew);
            float p1 = __expf(s1[r] - mnew);
            float rs = p0 + p1;
#pragma unroll
            for (int m = 1; m < 16; m <<= 1) rs += __shfl_xor(rs, m, 32);
            float corr = __expf(mstate[r] - mnew);
            lstate[r] = lstate[r] * corr + rs;
            mstate[r] = mnew;
            cf[r] = corr;
            s0[r] = p0; s1[r] = p1;
        }
#pragma unroll
        for (int j = 0; j < 16; ++j)
#pragma unroll
            for (int r = 0; r < 8; ++r) yacc[j][r] = yacc[j][r] * cf[r];

        // P (16x32) -> wave-private LDS (row-major) -> A fragment (2x b128)
        bf16* ps = pStage + wave * (16 * 32);
#pragma unroll
        for (int r = 0; r < 8; ++r) {
            ps[(r + 8 * h) * 32 + n]      = (bf16)s0[r];
            ps[(r + 8 * h) * 32 + 16 + n] = (bf16)s1[r];
        }
        v16bf pf = load_a_frag(ps, 32);
#pragma unroll
        for (int jg = 0; jg < 4; ++jg) {
            v16bf bg[4];
#pragma unroll
            for (int u = 0; u < 4; ++u) bg[u] = load_frag_sw(gp + (jg * 4 + u) * 512);
#pragma unroll
            for (int u = 0; u < 4; ++u)
                yacc[jg * 4 + u] = wmma_bf16(pf, bg[u], yacc[jg * 4 + u]);
        }
        p ^= 1;
    }

    // normalize and store y (B,T,D)
#pragma unroll
    for (int r = 0; r < 8; ++r) {
        float inv = 1.0f / lstate[r];
        int row = qrow0 + r + 8 * h;
#pragma unroll
        for (int j = 0; j < 16; ++j)
            yout[((size_t)b * Tn + row) * Dn + j * 16 + n] = (bf16)(yacc[j][r] * inv);
    }
}

// ---------------- GEMM-3: out = BN(W_w @ y^T + W_b) + x2 ----------------
// grid: (T/128, C/128, B), block 256. Double-buffered LDS.
__global__ void out_gemm_kernel(const bf16* __restrict__ wW,  // (C,D) bf16
                                const bf16* __restrict__ y,   // (B,T,D)
                                const float* __restrict__ W_b,
                                const float* __restrict__ bn_g,
                                const float* __restrict__ bn_b,
                                const float* __restrict__ bn_mean,
                                const float* __restrict__ bn_var,
                                const float* __restrict__ x2f,
                                float* __restrict__ out) {
    __shared__ __align__(32) bf16 At[2][128 * 32];
    __shared__ __align__(32) bf16 Bt[2][32 * 128];
    const int tn = blockIdx.x, tm = blockIdx.y, b = blockIdx.z;
    const int tid = threadIdx.x, wave = tid >> 5, lane = tid & 31;
    const int col0 = tn * 128, row0 = tm * 128;

    v8f acc[8] = {};
    int p = 0;
    auto stage = [&](int buf, int k0) {
        const bf16* Ag = wW + (size_t)row0 * Dn + k0;
        for (int i = tid; i < 512; i += 256) {
            int r = i >> 2, c8 = (i & 3) * 8;
            *(v8bf*)(At[buf] + r * 32 + c8) = *(const v8bf*)(Ag + (size_t)r * Dn + c8);
        }
        // B = y^T : y is (t, d) row-major, vector-load along d (=k)
        const bf16* Bg = y + ((size_t)b * Tn + col0) * Dn + k0;
        for (int i = tid; i < 512; i += 256) {
            int nn = i >> 2, k8 = (i & 3) * 8;
            v8bf v = *(const v8bf*)(Bg + (size_t)nn * Dn + k8);
#pragma unroll
            for (int e = 0; e < 8; ++e) Bt[buf][b_sw_idx(k8 + e, nn)] = v[e];
        }
    };
    stage(0, 0);
    for (int k0 = 0; k0 < Dn; k0 += 32) {
        __syncthreads();
        if (k0 + 32 < Dn) stage(p ^ 1, k0 + 32);
        v16bf a = load_a_frag(At[p] + wave * 16 * 32, 32);
        v16bf bfr[8];
#pragma unroll
        for (int j = 0; j < 8; ++j) bfr[j] = load_frag_sw(Bt[p] + j * 512);
#pragma unroll
        for (int j = 0; j < 8; ++j) acc[j] = wmma_bf16(a, bfr[j], acc[j]);
        p ^= 1;
    }
    const int h = lane >> 4, n = lane & 15;
#pragma unroll
    for (int j = 0; j < 8; ++j)
#pragma unroll
        for (int r = 0; r < 8; ++r) {
            int row = row0 + wave * 16 + r + 8 * h;   // channel o
            int col = col0 + j * 16 + n;              // t
            float scale = bn_g[row] * rsqrtf(bn_var[row] + EPSn);
            size_t idx = ((size_t)b * Cn + row) * Tn + col;
            float v = (acc[j][r] + W_b[row] - bn_mean[row]) * scale + bn_b[row] + x2f[idx];
            out[idx] = v;
        }
}

// ---------------- host launcher ----------------
extern "C" void kernel_launch(void* const* d_in, const int* in_sizes, int n_in,
                              void* d_out, int out_size, void* d_ws, size_t ws_size,
                              hipStream_t stream) {
    (void)in_sizes; (void)n_in; (void)out_size; (void)ws_size;
    const float* x      = (const float*)d_in[0];
    const float* tc_w1  = (const float*)d_in[1];
    const float* tc_w2  = (const float*)d_in[2];
    const float* z_w    = (const float*)d_in[3];
    const float* z_b    = (const float*)d_in[4];
    const float* g_w    = (const float*)d_in[5];
    const float* g_b    = (const float*)d_in[6];
    const float* th_w   = (const float*)d_in[7];
    const float* th_b   = (const float*)d_in[8];
    const float* ph_w   = (const float*)d_in[9];
    const float* ph_b   = (const float*)d_in[10];
    const float* W_w    = (const float*)d_in[11];
    const float* W_b    = (const float*)d_in[12];
    const float* bn_g   = (const float*)d_in[13];
    const float* bn_b   = (const float*)d_in[14];
    const float* bn_mean= (const float*)d_in[15];
    const float* bn_var = (const float*)d_in[16];

    const size_t nBCT = (size_t)Bn * Cn * Tn;   // 8,388,608
    const size_t nBTD = (size_t)Bn * Tn * Dn;   // 4,194,304

    char* p = (char*)d_ws;
    bf16*  xpe   = (bf16*)p;  p += nBCT * sizeof(bf16);
    float* x2f   = (float*)p; p += nBCT * sizeof(float);
    bf16*  x2h   = (bf16*)p;  p += nBCT * sizeof(bf16);
    bf16*  theta = (bf16*)p;  p += nBTD * sizeof(bf16);
    bf16*  phi   = (bf16*)p;  p += nBTD * sizeof(bf16);
    bf16*  gx    = (bf16*)p;  p += nBTD * sizeof(bf16);
    bf16*  yb    = (bf16*)p;  p += nBTD * sizeof(bf16);
    bf16*  Mall  = (bf16*)p;  p += (size_t)5 * Cn * Cn * sizeof(bf16);
    bf16*  wproj = (bf16*)p;  p += (size_t)3 * Dn * Cn * sizeof(bf16);
    bf16*  wWh   = (bf16*)p;  p += (size_t)Cn * Dn * sizeof(bf16);

    const int DC = Dn * Cn;  // 131072
    cvt_f32_bf16_kernel<<<(DC + 255) / 256, 256, 0, stream>>>(th_w, wproj + 0,       DC);
    cvt_f32_bf16_kernel<<<(DC + 255) / 256, 256, 0, stream>>>(ph_w, wproj + DC,      DC);
    cvt_f32_bf16_kernel<<<(DC + 255) / 256, 256, 0, stream>>>(g_w,  wproj + 2 * DC,  DC);
    cvt_f32_bf16_kernel<<<(DC + 255) / 256, 256, 0, stream>>>(W_w,  wWh,             DC);

    combine_w_kernel<<<(5 * Cn * Cn) / 256, 256, 0, stream>>>(z_w, tc_w1, tc_w2, Mall);

    pe_add_kernel<<<(int)(nBCT / 256), 256, 0, stream>>>(x, xpe);

    conv_z_gemm_kernel<<<dim3(Tn / 128, Cn / 128, Bn), 256, 0, stream>>>(
        Mall, xpe, z_b, x2f, x2h);

    proj_gemm_kernel<<<dim3(Tn / 128, Dn / 128, 3 * Bn), 256, 0, stream>>>(
        wproj, th_b, ph_b, g_b, x2h, theta, phi, gx);

    flash_attn_kernel<<<dim3(Tn / 128, Bn), 256, 0, stream>>>(theta, phi, gx, yb);

    out_gemm_kernel<<<dim3(Tn / 128, Cn / 128, Bn), 256, 0, stream>>>(
        wWh, yb, W_b, bn_g, bn_b, bn_mean, bn_var, x2f, (float*)d_out);
}
